// MultiHeadAttention_2568390443084
// MI455X (gfx1250) — matrix operand order repaired and checked by reading, not compiled, not run on hardware
//
#include <hip/hip_runtime.h>
#include <hip/hip_bf16.h>

// MHA forward, B=4, T=2048, D=1024, H=16, HD=64 on gfx1250 (wave32).
// All matmuls: v_wmma_f32_16x16x32_f16. GEMM staging: TDM tensor_load_to_lds
// (TENSORcnt double buffering) with async-copy fallback. Attention:
// flash-style, DPP ROW_XMASK row-max, WMMA-with-ones row-sums, mask only on
// the diagonal key block.
//
// Workspace (halves): xh 8M | WqT/WkT/WvT/WoT 1M each | Qh/Kh/Vt/ctx 8M each
// => 46,137,344 halves = 88 MB.

typedef __attribute__((ext_vector_type(16))) _Float16 v16h;
typedef __attribute__((ext_vector_type(8)))  _Float16 v8h;
typedef __attribute__((ext_vector_type(8)))  float    v8f;
typedef int gcc_v4i __attribute__((vector_size(16)));
typedef unsigned tu4 __attribute__((vector_size(16)));
typedef int tv4 __attribute__((vector_size(16)));
typedef int tv8 __attribute__((vector_size(32)));

#define DEV static __device__ __forceinline__

#if (__has_builtin(__builtin_amdgcn_global_load_async_to_lds_b128) && \
     __has_builtin(__builtin_amdgcn_s_wait_asynccnt))
#define HAS_ASYNC_COPY 1
#else
#define HAS_ASYNC_COPY 0
#endif

#if (__has_builtin(__builtin_amdgcn_tensor_load_to_lds) && \
     __has_builtin(__builtin_amdgcn_s_wait_tensorcnt))
#define HAS_TDM 1
#else
#define HAS_TDM 0
#endif

// Copy 16 bytes global -> LDS (ASYNCcnt-tracked when available).
DEV void cp16(_Float16* l, const _Float16* g) {
#if HAS_ASYNC_COPY
  typedef __attribute__((address_space(1))) gcc_v4i GV4;
  typedef __attribute__((address_space(3))) gcc_v4i LV4;
  __builtin_amdgcn_global_load_async_to_lds_b128((GV4*)g, (LV4*)l, 0, 0);
#else
  *(v8h*)l = *(const v8h*)g;
#endif
}

#if HAS_TDM
DEV unsigned lds_addr32(const void* p) {
  typedef __attribute__((address_space(3))) char LCH;
  return (unsigned)(size_t)(LCH*)p;
}

// One TDM descriptor: load a [rows x row_halves] f16 tile (global row stride
// ld_halves) into contiguous LDS at lds_off. D# layout per ISA 8.3/8.4.
DEV void tdm_load_2d(const _Float16* g, unsigned lds_off,
                     int rows, int row_halves, int ld_halves) {
  unsigned long long ga = (unsigned long long)(size_t)g;
  tu4 g0;
  g0[0] = 1u;                                  // count=1, user descriptor
  g0[1] = lds_off;                             // lds_addr (bytes)
  g0[2] = (unsigned)ga;                        // global_addr[31:0]
  g0[3] = ((unsigned)(ga >> 32) & 0x01FFFFFFu) | (2u << 30); // [56:32]|type=2
  tv8 g1;
  g1[0] = 1 << 16;                             // data_size=1 (2-byte elements)
  g1[1] = (int)(0xFFFFu << 16);                // tensor_dim0 = 0x7FFFFFFF (lo16)
  g1[2] = (int)(0x7FFFu | (0xFFFFu << 16));    // dim0 hi16 | tensor_dim1 lo16
  g1[3] = (int)(0x7FFFu | ((unsigned)row_halves << 16)); // dim1 hi16 | tile_dim0
  g1[4] = rows;                                // tile_dim1 (tile_dim2 = 0)
  g1[5] = ld_halves;                           // tensor_dim0_stride lo32
  g1[6] = 0;                                   // stride hi16 | dim1_stride lo16
  g1[7] = 0;
  tv4 z4 = {0, 0, 0, 0};
  tv8 z8 = {0, 0, 0, 0, 0, 0, 0, 0};
  __builtin_amdgcn_tensor_load_to_lds(g0, g1, z4, z4, z8, 0);
}
#endif

// Max across the 16-lane half-wave via DPP16 ROW_XMASK (VALU, no LDS pipe).
DEV float dpp_max16(float x) {
  x = fmaxf(x, __int_as_float(__builtin_amdgcn_update_dpp(
                  0, __float_as_int(x), 0x161, 0xf, 0xf, true)));
  x = fmaxf(x, __int_as_float(__builtin_amdgcn_update_dpp(
                  0, __float_as_int(x), 0x162, 0xf, 0xf, true)));
  x = fmaxf(x, __int_as_float(__builtin_amdgcn_update_dpp(
                  0, __float_as_int(x), 0x164, 0xf, 0xf, true)));
  x = fmaxf(x, __int_as_float(__builtin_amdgcn_update_dpp(
                  0, __float_as_int(x), 0x168, 0xf, 0xf, true)));
  return x;
}

// A-matrix 16x32 f16 fragment (ISA 7.12.2): lane L holds row m=L%16,
// K = {0..7,16..23} (lanes 0-15) / {8..15,24..31} (lanes 16-31).
DEV v16h load_a_frag(const _Float16* base, int ldk) {
  int lane = threadIdx.x & 31;
  int m = lane & 15, hi = lane >> 4;
  const _Float16* p = base + (size_t)m * ldk + hi * 8;
  v8h lo = *(const v8h*)(p);
  v8h hh = *(const v8h*)(p + 16);
  v16h r;
#pragma unroll
  for (int i = 0; i < 8; ++i) { r[i] = lo[i]; r[i + 8] = hh[i]; }
  return r;
}

// B-matrix 32x16 f16 fragment: lane L holds column n=L%16,
// K = {0..7,16..23} (lanes 0-15) / {8..15,24..31} (lanes 16-31).
DEV v16h load_b_frag(const _Float16* bt, int ldk) {
  int lane = threadIdx.x & 31;
  int n = lane & 15, hi = lane >> 4;
  const _Float16* p = bt + (size_t)n * ldk + hi * 8;
  v8h lo = *(const v8h*)(p);
  v8h hh = *(const v8h*)(p + 16);
  v16h r;
#pragma unroll
  for (int i = 0; i < 8; ++i) { r[i] = lo[i]; r[i + 8] = hh[i]; }
  return r;
}

DEV v8f wmma16(v16h a, v16h b, v8f c) {
  return __builtin_amdgcn_wmma_f32_16x16x32_f16(false, a, false, b,
                                                (short)0, c, false, false);
}

// ---------------------------------------------------------------- converts
__global__ __launch_bounds__(256) void cvt_f16_kernel(const float* __restrict__ s,
                                                      _Float16* __restrict__ d, int n) {
  int i = blockIdx.x * 256 + threadIdx.x;
  if (i < n) d[i] = (_Float16)s[i];
}

__global__ __launch_bounds__(256) void transpose_cvt_kernel(const float* __restrict__ W,
                                                            _Float16* __restrict__ Wt,
                                                            int K, int N) {
  __shared__ float tile[16][17];
  int n0 = blockIdx.x * 16, k0 = blockIdx.y * 16;
  int tn = threadIdx.x & 15, tk = threadIdx.x >> 4;
  tile[tk][tn] = W[(size_t)(k0 + tk) * N + (n0 + tn)];
  __syncthreads();
  int wk = threadIdx.x & 15, wn = threadIdx.x >> 4;
  Wt[(size_t)(n0 + wn) * K + (k0 + wk)] = (_Float16)tile[wk][wn];
}

// ------------------------------------------------------------------- GEMM
// Fallback staging: 64x32 A tile + 256x32 B^T tile (5 x 16B per thread).
DEV void stage_tile(_Float16* As, _Float16* Bs, const _Float16* __restrict__ A,
                    const _Float16* __restrict__ Bt, int m0, int n0, int k, int K,
                    int t) {
  int arow = t >> 2, aoff = (t & 3) * 8;
  cp16(As + arow * 32 + aoff, A + (size_t)(m0 + arow) * K + k + aoff);
#pragma unroll
  for (int i = 0; i < 4; ++i) {
    int c = t + i * 256, brow = c >> 2, boff = (c & 3) * 8;
    cp16(Bs + brow * 32 + boff, Bt + (size_t)(n0 + brow) * K + k + boff);
  }
}

// Block: 256 threads (8 waves), tile 64(M) x 256(N), K-step 32, double-
// buffered LDS staging (TDM preferred). Wave (2x4) computes 32x64.
// MODE 0: C f16 = scale*A@B; MODE 1: f16 scatter Vt[b,h,d,t]; MODE 2: f32+bias
template <int MODE>
__global__ __launch_bounds__(256) void gemm_kernel(const _Float16* __restrict__ A,
                                                   const _Float16* __restrict__ Bt,
                                                   _Float16* __restrict__ Ch,
                                                   float* __restrict__ Cf,
                                                   const float* __restrict__ bias,
                                                   int M, int N, int K, float scale,
                                                   int T, int H) {
  __shared__ _Float16 As[2][64 * 32];
  __shared__ _Float16 Bs[2][256 * 32];
  const int t = threadIdx.x;
  const int wave = t >> 5;
  const int m0 = blockIdx.x * 64;
  const int n0 = blockIdx.y * 256;
  const int wm = (wave >> 2) * 32;
  const int wn = (wave & 3) * 64;

  v8f acc[2][4];
#pragma unroll
  for (int f = 0; f < 2; ++f)
#pragma unroll
    for (int g = 0; g < 4; ++g) acc[f][g] = v8f{};

  const int nk = K >> 5;

#if HAS_TDM
  const bool issuer = (wave == 0);
  unsigned asb[2] = {lds_addr32(&As[0][0]), lds_addr32(&As[1][0])};
  unsigned bsb[2] = {lds_addr32(&Bs[0][0]), lds_addr32(&Bs[1][0])};
  if (issuer) {
    tdm_load_2d(A + (size_t)m0 * K, asb[0], 64, 32, K);
    tdm_load_2d(Bt + (size_t)n0 * K, bsb[0], 256, 32, K);
  }
#else
  stage_tile(&As[0][0], &Bs[0][0], A, Bt, m0, n0, 0, K, t);
#endif

  for (int kk = 0; kk < nk; ++kk) {
    const int cur = kk & 1;
    const bool has_next = (kk + 1 < nk);
#if HAS_TDM
    if (issuer) {
      if (has_next) {   // 2 TDM ops per tile, in-order: <=2 => current landed
        int k = (kk + 1) << 5;
        tdm_load_2d(A + (size_t)m0 * K + k, asb[cur ^ 1], 64, 32, K);
        tdm_load_2d(Bt + (size_t)n0 * K + k, bsb[cur ^ 1], 256, 32, K);
        __builtin_amdgcn_s_wait_tensorcnt(2);
      } else {
        __builtin_amdgcn_s_wait_tensorcnt(0);
      }
    }
#else
    if (has_next)
      stage_tile(&As[cur ^ 1][0], &Bs[cur ^ 1][0], A, Bt, m0, n0, (kk + 1) << 5, K, t);
#if HAS_ASYNC_COPY
    if (has_next) __builtin_amdgcn_s_wait_asynccnt(5);
    else          __builtin_amdgcn_s_wait_asynccnt(0);
#endif
#endif
    __syncthreads();   // current tile visible to all waves

    v16h a0 = load_a_frag(&As[cur][(wm + 0) * 32], 32);
    v16h a1 = load_a_frag(&As[cur][(wm + 16) * 32], 32);
#pragma unroll
    for (int g = 0; g < 4; ++g) {
      v16h b = load_b_frag(&Bs[cur][(wn + g * 16) * 32], 32);
      acc[0][g] = wmma16(a0, b, acc[0][g]);
      acc[1][g] = wmma16(a1, b, acc[1][g]);
    }
    __syncthreads();   // all waves done reading before refill
  }

  const int lane = t & 31, nl = lane & 15, hi = lane >> 4;
#pragma unroll
  for (int f = 0; f < 2; ++f)
#pragma unroll
    for (int g = 0; g < 4; ++g)
#pragma unroll
      for (int v = 0; v < 8; ++v) {
        int m = m0 + wm + f * 16 + v + 8 * hi;
        int n = n0 + wn + g * 16 + nl;
        if (MODE == 0) {
          Ch[(size_t)m * N + n] = (_Float16)(acc[f][g][v] * scale);
        } else if (MODE == 1) {
          const int HD = 64;
          int bb = m / T, tt = m - bb * T;
          int hh = n >> 6, dd = n & 63;
          Ch[(((size_t)bb * H + hh) * HD + dd) * (size_t)T + tt] =
              (_Float16)(acc[f][g][v] * scale);
        } else {
          Cf[(size_t)m * N + n] = acc[f][g][v] + bias[n];
        }
      }
}

// -------------------------------------------------------- flash attention
// One wave per (b, h, 32-query block); 32-key blocks, online softmax.
// Interior blocks skip the causal mask; only the diagonal block masks.
__global__ __launch_bounds__(32) void attn_kernel(const _Float16* __restrict__ Q,
                                                  const _Float16* __restrict__ Kmat,
                                                  const _Float16* __restrict__ Vt,
                                                  _Float16* __restrict__ Ctx,
                                                  int B, int T, int H) {
  const int HD = 64;
  const int D = H * HD;
  __shared__ _Float16 Plds[32 * 32];

  const int q0 = blockIdx.x * 32;
  const int h  = blockIdx.y;
  const int b  = blockIdx.z;
  const int lane = threadIdx.x & 31;
  const int nl = lane & 15, hi = lane >> 4;

  const _Float16* qbase = Q + ((size_t)b * T + q0) * D + h * HD;
  v16h qf[2][2];
#pragma unroll
  for (int r = 0; r < 2; ++r)
#pragma unroll
    for (int c = 0; c < 2; ++c)
      qf[r][c] = load_a_frag(qbase + (size_t)(16 * r) * D + 32 * c, D);

  v16h onesb;
#pragma unroll
  for (int i = 0; i < 16; ++i) onesb[i] = (_Float16)1.0f;

  float mrun[2][8], lrun[2][8];
  v8f o[2][4];
#pragma unroll
  for (int r = 0; r < 2; ++r)
#pragma unroll
    for (int v = 0; v < 8; ++v) { mrun[r][v] = -3.0e38f; lrun[r][v] = 0.f; }
#pragma unroll
  for (int r = 0; r < 2; ++r)
#pragma unroll
    for (int g = 0; g < 4; ++g) o[r][g] = v8f{};

  const _Float16* vbase0 = Vt + (((size_t)b * H + h) * HD) * (size_t)T;

  auto process_block = [&](int k0, bool domask) {
    const _Float16* kbase = Kmat + ((size_t)b * T + k0) * D + h * HD;
    if (k0 < q0) {                            // hint next key block
      __builtin_prefetch(kbase + (size_t)32 * D, 0, 0);
      __builtin_prefetch(vbase0 + k0 + 32, 0, 0);
    }

    // scores for 32 q x 32 keys: 4 K fragments, 8 WMMAs
    v16h kb00 = load_b_frag(kbase, D);
    v16h kb01 = load_b_frag(kbase + 32, D);
    v16h kb10 = load_b_frag(kbase + (size_t)16 * D, D);
    v16h kb11 = load_b_frag(kbase + (size_t)16 * D + 32, D);
    v8f s[2][2];
#pragma unroll
    for (int r = 0; r < 2; ++r) {
      s[r][0] = wmma16(qf[r][0], kb00, v8f{});
      s[r][0] = wmma16(qf[r][1], kb01, s[r][0]);
      s[r][1] = wmma16(qf[r][0], kb10, v8f{});
      s[r][1] = wmma16(qf[r][1], kb11, s[r][1]);
    }

    if (domask) {   // only the diagonal block needs the causal mask
#pragma unroll
      for (int r = 0; r < 2; ++r)
#pragma unroll
        for (int v = 0; v < 8; ++v) {
          int q = q0 + 16 * r + v + 8 * hi;
          if (k0 + nl > q)      s[r][0][v] = -3.0e38f;
          if (k0 + 16 + nl > q) s[r][1][v] = -3.0e38f;
        }
    }

    // online softmax: row max via DPP butterfly
    float rescale[2][8];
#pragma unroll
    for (int r = 0; r < 2; ++r)
#pragma unroll
      for (int v = 0; v < 8; ++v) {
        float mx = dpp_max16(fmaxf(s[r][0][v], s[r][1][v]));
        float mnew = fmaxf(mrun[r][v], mx);
        rescale[r][v] = __expf(mrun[r][v] - mnew);
        mrun[r][v] = mnew;
      }
#pragma unroll
    for (int r = 0; r < 2; ++r)
#pragma unroll
      for (int v = 0; v < 8; ++v) {
        s[r][0][v] = __expf(s[r][0][v] - mrun[r][v]);
        s[r][1][v] = __expf(s[r][1][v] - mrun[r][v]);
#pragma unroll
        for (int g = 0; g < 4; ++g) o[r][g][v] *= rescale[r][v];
      }

    // route P (C layout) -> A-fragment layout through LDS
    __syncthreads();
#pragma unroll
    for (int r = 0; r < 2; ++r)
#pragma unroll
      for (int v = 0; v < 8; ++v) {
        int row = 16 * r + v + 8 * hi;
        Plds[row * 32 + nl]      = (_Float16)s[r][0][v];
        Plds[row * 32 + 16 + nl] = (_Float16)s[r][1][v];
      }
    __syncthreads();
    v16h pf0 = load_a_frag(Plds, 32);
    v16h pf1 = load_a_frag(Plds + 16 * 32, 32);

    // row sums via WMMA against all-ones B: C[m][*] = rowsum(P[m])
    v8f sum0 = wmma16(pf0, onesb, v8f{});
    v8f sum1 = wmma16(pf1, onesb, v8f{});
#pragma unroll
    for (int v = 0; v < 8; ++v) {
      lrun[0][v] = lrun[0][v] * rescale[0][v] + sum0[v];
      lrun[1][v] = lrun[1][v] * rescale[1][v] + sum1[v];
    }

    // O += P @ V : 4 V fragments, 8 WMMAs
    const _Float16* vbase = vbase0 + k0;
#pragma unroll
    for (int g = 0; g < 4; ++g) {
      v16h vb = load_b_frag(vbase + (size_t)(g * 16) * T, T);
      o[0][g] = wmma16(pf0, vb, o[0][g]);
      o[1][g] = wmma16(pf1, vb, o[1][g]);
    }
  };

  const int ndiag = q0 >> 5;                 // fully-unmasked key blocks
  for (int kb = 0; kb < ndiag; ++kb) process_block(kb * 32, false);
  process_block(q0, true);                   // diagonal block

  // epilogue: O /= l, store ctx [B,T,D] f16
  _Float16* cb = Ctx + ((size_t)b * T + q0) * D + h * HD;
#pragma unroll
  for (int r = 0; r < 2; ++r)
#pragma unroll
    for (int v = 0; v < 8; ++v) {
      float inv = 1.0f / lrun[r][v];
      int row = 16 * r + v + 8 * hi;
#pragma unroll
      for (int g = 0; g < 4; ++g)
        cb[(size_t)row * D + g * 16 + nl] = (_Float16)(o[r][g][v] * inv);
    }
}

// ------------------------------------------------------------------ launch
extern "C" void kernel_launch(void* const* d_in, const int* in_sizes, int n_in,
                              void* d_out, int out_size, void* d_ws, size_t ws_size,
                              hipStream_t stream) {
  const int B = 4, T = 2048, DIN = 1024, DOUT = 1024, H = 16;
  const int M = B * T;                 // 8192
  const float qscale = 0.125f;         // 1/sqrt(HD=64)

  const float* x  = (const float*)d_in[0];
  const float* Wq = (const float*)d_in[1];
  const float* Wk = (const float*)d_in[2];
  const float* Wv = (const float*)d_in[3];
  const float* Wo = (const float*)d_in[4];
  const float* bo = (const float*)d_in[5];
  float* out = (float*)d_out;

  _Float16* ws = (_Float16*)d_ws;
  _Float16* xh  = ws;                     size_t off = (size_t)M * DIN;
  _Float16* wqT = ws + off;               off += (size_t)DIN * DOUT;
  _Float16* wkT = ws + off;               off += (size_t)DIN * DOUT;
  _Float16* wvT = ws + off;               off += (size_t)DIN * DOUT;
  _Float16* woT = ws + off;               off += (size_t)DOUT * DOUT;
  _Float16* Qh  = ws + off;               off += (size_t)M * DOUT;
  _Float16* Kh  = ws + off;               off += (size_t)M * DOUT;
  _Float16* Vth = ws + off;               off += (size_t)M * DOUT;
  _Float16* ctx = ws + off;               off += (size_t)M * DOUT;
  // requires ws_size >= off * 2 bytes (88 MB)

  // 1) casts / transposes
  int nx = M * DIN;
  cvt_f16_kernel<<<(nx + 255) / 256, 256, 0, stream>>>(x, xh, nx);
  dim3 tg(DOUT / 16, DIN / 16);
  transpose_cvt_kernel<<<tg, 256, 0, stream>>>(Wq, wqT, DIN, DOUT);
  transpose_cvt_kernel<<<tg, 256, 0, stream>>>(Wk, wkT, DIN, DOUT);
  transpose_cvt_kernel<<<tg, 256, 0, stream>>>(Wv, wvT, DIN, DOUT);
  transpose_cvt_kernel<<<dim3(DOUT / 16, DOUT / 16), 256, 0, stream>>>(Wo, woT, DOUT, DOUT);

  // 2) projections (TDM/async double-buffered WMMA GEMMs)
  dim3 gg(M / 64, DOUT / 256);
  gemm_kernel<0><<<gg, 256, 0, stream>>>(xh, wqT, Qh,  nullptr, nullptr,
                                         M, DOUT, DIN, qscale, T, H);
  gemm_kernel<0><<<gg, 256, 0, stream>>>(xh, wkT, Kh,  nullptr, nullptr,
                                         M, DOUT, DIN, 1.0f, T, H);
  gemm_kernel<1><<<gg, 256, 0, stream>>>(xh, wvT, Vth, nullptr, nullptr,
                                         M, DOUT, DIN, 1.0f, T, H);

  // 3) streaming causal attention
  attn_kernel<<<dim3(T / 32, H, B), 32, 0, stream>>>(Qh, Kh, Vth, ctx, B, T, H);

  // 4) output projection + bias (f32 out)
  gemm_kernel<2><<<gg, 256, 0, stream>>>(ctx, woT, nullptr, out, bo,
                                         M, DOUT, DOUT, 1.0f, T, H);
}
